// ProtoPNetDist_33139967656189
// MI455X (gfx1250) — matrix operand order repaired and checked by reading, not compiled, not run on hardware
//
#include <hip/hip_runtime.h>
#include <math.h>

// Problem constants (from reference): B,S,E, P,K, C
#define B_ 64
#define S_ 512
#define E_ 1024
#define P_ 200
#define K_ 10
#define C_ 3

typedef __attribute__((ext_vector_type(16))) __bf16 v16bf;
typedef __attribute__((ext_vector_type(8)))  __bf16 v8bf;
typedef __attribute__((ext_vector_type(4)))  __bf16 v4bf;
typedef __attribute__((ext_vector_type(8)))  float  v8f;

// Split fp32 into bf16 hi + bf16 lo (x ~= hi + lo); bf16 products are exact in
// the WMMA multiplier, so Ahi*Bhi + Ahi*Blo + Alo*Bhi ~ fp32-accurate xy.
__device__ __forceinline__ void split_bf16(float f, __bf16& hi, __bf16& lo) {
  hi = (__bf16)f;
  lo = (__bf16)(f - (float)hi);
}

// ---------------------------------------------------------------------------
// Kernel 0: pre-split an fp32 array into bf16 hi/lo planes (4 elems/thread).
// Hoists all conversion VALU out of the WMMA inner loop.
// ---------------------------------------------------------------------------
__global__ __launch_bounds__(256)
void pp_split_planes(const float* __restrict__ x, __bf16* __restrict__ hi,
                     __bf16* __restrict__ lo, int n4) {
  const int i = blockIdx.x * blockDim.x + threadIdx.x;
  if (i >= n4) return;
  float4 v = ((const float4*)x)[i];
  __bf16 h0, h1, h2, h3, l0, l1, l2, l3;
  split_bf16(v.x, h0, l0);
  split_bf16(v.y, h1, l1);
  split_bf16(v.z, h2, l2);
  split_bf16(v.w, h3, l3);
  ((v4bf*)hi)[i] = (v4bf){h0, h1, h2, h3};
  ((v4bf*)lo)[i] = (v4bf){l0, l1, l2, l3};
}

// ---------------------------------------------------------------------------
// Kernel 1: per-row sum of squares (x2 [B*S], y2 [P*K]); one wave per row.
// ---------------------------------------------------------------------------
__global__ __launch_bounds__(256)
void pp_row_sumsq(const float* __restrict__ x, float* __restrict__ out, int nrows) {
  const int wave = threadIdx.x >> 5;
  const int lane = threadIdx.x & 31;
  const int r = blockIdx.x * 8 + wave;
  if (r >= nrows) return;
  const float4* row = (const float4*)(x + (size_t)r * E_);
  float s = 0.f;
#pragma unroll
  for (int i = 0; i < 8; ++i) {
    float4 v = row[lane + i * 32];
    s += v.x * v.x + v.y * v.y + v.z * v.z + v.w * v.w;
  }
#pragma unroll
  for (int off = 16; off > 0; off >>= 1) s += __shfl_xor(s, off, 32);
  if (lane == 0) out[r] = s;
}

// ---------------------------------------------------------------------------
// Kernel 2: initialize per-(b,p,k) running-min buffer to +inf (ordered uint).
// ---------------------------------------------------------------------------
__global__ void pp_init_min(unsigned int* __restrict__ p, int n) {
  int i = blockIdx.x * blockDim.x + threadIdx.x;
  if (i < n) p[i] = 0x7F800000u;
}

// ---------------------------------------------------------------------------
// Shared epilogue: d = sqrt(max(x2+y2-2*xy,0)), scatter-store, min over s.
// ---------------------------------------------------------------------------
__device__ __forceinline__
void pp_epilogue(v8f acc, int m_base, int n_lane, int b, int s_b, int hi16,
                 const float* __restrict__ x2, const float* __restrict__ y2,
                 float* __restrict__ dist_out, unsigned int* __restrict__ minbuf) {
  const int   p   = n_lane / K_;
  const int   k   = n_lane - p * K_;
  const float y2n = y2[n_lane];
  float mn = 3.402823466e38f;
#pragma unroll
  for (int r = 0; r < 8; ++r) {
    const int   row = r + hi16 * 8;  // C/D layout: M = r + 8*hi16, N = lane&15
    const int   s   = s_b + row;
    const float d2  = x2[m_base + row] + y2n - 2.0f * acc[r];
    const float d   = sqrtf(fmaxf(d2, 0.0f));
    dist_out[(((size_t)b * P_ + p) * S_ + s) * K_ + k] = d;
    mn = fminf(mn, d);
  }
  mn = fminf(mn, __shfl_xor(mn, 16, 32));  // lanes l, l+16 share n
  if (hi16 == 0) {
    atomicMin(minbuf + (size_t)(b * P_ + p) * K_ + k, __float_as_uint(mn));
  }
}

// ---------------------------------------------------------------------------
// Kernel 3a (fast path): GEMM from pre-split bf16 planes. Inner loop is
// 8x global_load_b128 + 3x v_wmma — WMMA-issue-bound, ~no VALU.
// Grid: x = N tiles (125)  [fast-varying -> concurrent blocks share A slab],
//       y = M slabs (256 of 128 rows). Block = 8 waves, one 16x16 tile each.
// ---------------------------------------------------------------------------
__global__ __launch_bounds__(256)
void pp_gemm_dist_planes(const __bf16* __restrict__ ehi, const __bf16* __restrict__ elo,
                         const __bf16* __restrict__ phi, const __bf16* __restrict__ plo,
                         const float* __restrict__ x2, const float* __restrict__ y2,
                         float* __restrict__ dist_out, unsigned int* __restrict__ minbuf) {
  const int lane = threadIdx.x & 31;
  const int wave = threadIdx.x >> 5;
  const int col  = lane & 15;
  const int hi16 = lane >> 4;

  const int m_base = (blockIdx.y * 8 + wave) * 16;  // row over B*S
  const int n_base = blockIdx.x * 16;               // col over P*K
  const int b      = m_base / S_;
  const int s_b    = m_base - b * S_;

  // A (16x32 bf16): lane holds row (lane&15); element j -> k = j+16*hi16
  //   -> two contiguous 8-elem (16B) runs at k0 and k0+8.
  const size_t arow = (size_t)(m_base + col) * E_;
  const int    ak   = hi16 * 16;
  // B (32x16 bf16): lane holds column n_base+(lane&15); element j -> k = j+16*hi16
  //   -> one contiguous 16-elem (32B) run.
  const int    n_lane = n_base + col;
  const size_t brow   = (size_t)n_lane * E_;
  const int    bk     = hi16 * 16;

  const __bf16* ah = ehi + arow + ak;
  const __bf16* al = elo + arow + ak;
  const __bf16* bh = phi + brow + bk;
  const __bf16* bl = plo + brow + bk;

  v8f acc = {};
  for (int e0 = 0; e0 < E_; e0 += 32) {
    v8bf ah0 = *(const v8bf*)(ah + e0);
    v8bf ah1 = *(const v8bf*)(ah + e0 + 8);
    v8bf al0 = *(const v8bf*)(al + e0);
    v8bf al1 = *(const v8bf*)(al + e0 + 8);
    v8bf bh0 = *(const v8bf*)(bh + e0);
    v8bf bh1 = *(const v8bf*)(bh + e0 + 8);
    v8bf bl0 = *(const v8bf*)(bl + e0);
    v8bf bl1 = *(const v8bf*)(bl + e0 + 8);
    __builtin_prefetch(ah + e0 + 64, 0, 0);  // streaming A planes
    __builtin_prefetch(al + e0 + 64, 0, 0);

    v16bf ahi = __builtin_shufflevector(ah0, ah1, 0,1,2,3,4,5,6,7,8,9,10,11,12,13,14,15);
    v16bf alo = __builtin_shufflevector(al0, al1, 0,1,2,3,4,5,6,7,8,9,10,11,12,13,14,15);
    v16bf bhi = __builtin_shufflevector(bh0, bh1, 0,1,2,3,4,5,6,7,8,9,10,11,12,13,14,15);
    v16bf blo = __builtin_shufflevector(bl0, bl1, 0,1,2,3,4,5,6,7,8,9,10,11,12,13,14,15);

    acc = __builtin_amdgcn_wmma_f32_16x16x32_bf16(false, ahi, false, bhi,
                                                  (short)0, acc, false, false);
    acc = __builtin_amdgcn_wmma_f32_16x16x32_bf16(false, ahi, false, blo,
                                                  (short)0, acc, false, false);
    acc = __builtin_amdgcn_wmma_f32_16x16x32_bf16(false, alo, false, bhi,
                                                  (short)0, acc, false, false);
  }
  pp_epilogue(acc, m_base, n_lane, b, s_b, hi16, x2, y2, dist_out, minbuf);
}

// ---------------------------------------------------------------------------
// Kernel 3b (fallback if workspace too small for planes): convert in-loop.
// ---------------------------------------------------------------------------
__global__ __launch_bounds__(256)
void pp_gemm_dist_conv(const float* __restrict__ emb, const float* __restrict__ proto,
                       const float* __restrict__ x2, const float* __restrict__ y2,
                       float* __restrict__ dist_out, unsigned int* __restrict__ minbuf) {
  const int lane = threadIdx.x & 31;
  const int wave = threadIdx.x >> 5;
  const int col  = lane & 15;
  const int hi16 = lane >> 4;

  const int m_base = (blockIdx.y * 8 + wave) * 16;
  const int n_base = blockIdx.x * 16;
  const int b      = m_base / S_;
  const int s_b    = m_base - b * S_;

  const float* arow  = emb + (size_t)(m_base + col) * E_;
  const int    abase = hi16 * 16;
  const int    n_lane = n_base + col;
  const float* brow   = proto + (size_t)n_lane * E_;
  const int    bbase  = hi16 * 16;

  v8f acc = {};
  for (int e0 = 0; e0 < E_; e0 += 32) {
    float af[16], bfv[16];
    {
      const float4* pa = (const float4*)(arow + e0 + abase);
      const float4* pb = (const float4*)(arow + e0 + abase + 8);
      float4 t0 = pa[0], t1 = pa[1], t2 = pb[0], t3 = pb[1];
      af[0]=t0.x; af[1]=t0.y; af[2]=t0.z; af[3]=t0.w;
      af[4]=t1.x; af[5]=t1.y; af[6]=t1.z; af[7]=t1.w;
      af[8]=t2.x; af[9]=t2.y; af[10]=t2.z; af[11]=t2.w;
      af[12]=t3.x; af[13]=t3.y; af[14]=t3.z; af[15]=t3.w;
      const float4* pc = (const float4*)(brow + e0 + bbase);
      float4 u0 = pc[0], u1 = pc[1], u2 = pc[2], u3 = pc[3];
      bfv[0]=u0.x; bfv[1]=u0.y; bfv[2]=u0.z; bfv[3]=u0.w;
      bfv[4]=u1.x; bfv[5]=u1.y; bfv[6]=u1.z; bfv[7]=u1.w;
      bfv[8]=u2.x; bfv[9]=u2.y; bfv[10]=u2.z; bfv[11]=u2.w;
      bfv[12]=u3.x; bfv[13]=u3.y; bfv[14]=u3.z; bfv[15]=u3.w;
    }
    v16bf ahi, alo, bhi, blo;
#pragma unroll
    for (int j = 0; j < 16; ++j) {
      __bf16 h, l;
      split_bf16(af[j], h, l);  ahi[j] = h; alo[j] = l;
      split_bf16(bfv[j], h, l); bhi[j] = h; blo[j] = l;
    }
    acc = __builtin_amdgcn_wmma_f32_16x16x32_bf16(false, ahi, false, bhi,
                                                  (short)0, acc, false, false);
    acc = __builtin_amdgcn_wmma_f32_16x16x32_bf16(false, ahi, false, blo,
                                                  (short)0, acc, false, false);
    acc = __builtin_amdgcn_wmma_f32_16x16x32_bf16(false, alo, false, bhi,
                                                  (short)0, acc, false, false);
  }
  pp_epilogue(acc, m_base, n_lane, b, s_b, hi16, x2, y2, dist_out, minbuf);
}

// ---------------------------------------------------------------------------
// Kernel 4: mean over K of min buffer -> pd [B,P]; class_out = pd @ fc^T.
// ---------------------------------------------------------------------------
__global__ __launch_bounds__(256)
void pp_finalize(const unsigned int* __restrict__ minbuf, const float* __restrict__ fc,
                 float* __restrict__ out_pd, float* __restrict__ out_cls) {
  __shared__ float pd[P_];
  const int b = blockIdx.x;
  const int t = threadIdx.x;
  if (t < P_) {
    const unsigned int* mrow = minbuf + (size_t)(b * P_ + t) * K_;
    float s = 0.f;
#pragma unroll
    for (int k = 0; k < K_; ++k) s += __uint_as_float(mrow[k]);
    s *= (1.0f / K_);
    pd[t] = s;
    out_pd[b * P_ + t] = s;
  }
  __syncthreads();
  if (t < C_) {
    float s = 0.f;
    for (int p = 0; p < P_; ++p) s += pd[p] * fc[t * P_ + p];
    out_cls[b * C_ + t] = s;
  }
}

// ---------------------------------------------------------------------------
extern "C" void kernel_launch(void* const* d_in, const int* in_sizes, int n_in,
                              void* d_out, int out_size, void* d_ws, size_t ws_size,
                              hipStream_t stream) {
  const float* emb   = (const float*)d_in[0];  // [B,S,E]
  const float* proto = (const float*)d_in[1];  // [P,K,E]
  const float* fc    = (const float*)d_in[2];  // [C,P]

  float* out_pd   = (float*)d_out;
  float* out_dist = out_pd + (size_t)B_ * P_;
  float* out_cls  = out_dist + (size_t)B_ * P_ * S_ * K_;

  // Workspace: [x2 | y2 | minbuf | (optional) emb_hi | emb_lo | proto_hi | proto_lo]
  char* ws = (char*)d_ws;
  float*        x2     = (float*)ws;                                           // 128KB
  float*        y2     = (float*)(ws + (size_t)(B_ * S_) * sizeof(float));     // 8KB
  unsigned int* minbuf = (unsigned int*)(ws + (size_t)(B_ * S_ + 2048) * sizeof(float));
  const int    nmin   = B_ * P_ * K_;            // 128000
  const size_t nEmb   = (size_t)B_ * S_ * E_;    // 33,554,432
  const size_t nProto = (size_t)P_ * K_ * E_;    // 2,048,000

  const size_t planesOff = 1u << 20;  // 1 MB: covers x2/y2/minbuf, keeps alignment
  __bf16* ehi = (__bf16*)(ws + planesOff);
  __bf16* elo = ehi + nEmb;
  __bf16* phi = elo + nEmb;
  __bf16* plo = phi + nProto;
  const size_t needPlanes = planesOff + 2 * (nEmb + nProto) * sizeof(__bf16);

  hipLaunchKernelGGL(pp_init_min, dim3((nmin + 255) / 256), dim3(256), 0, stream,
                     minbuf, nmin);
  hipLaunchKernelGGL(pp_row_sumsq, dim3((B_ * S_) / 8), dim3(256), 0, stream,
                     emb, x2, B_ * S_);
  hipLaunchKernelGGL(pp_row_sumsq, dim3((P_ * K_) / 8), dim3(256), 0, stream,
                     proto, y2, P_ * K_);

  const dim3 gemmGrid((P_ * K_) / 16, (B_ * S_) / 128);  // x = N tiles (L2 reuse of A)
  if (ws_size >= needPlanes) {
    hipLaunchKernelGGL(pp_split_planes, dim3((unsigned)(nEmb / 4 / 256)), dim3(256),
                       0, stream, emb, ehi, elo, (int)(nEmb / 4));
    hipLaunchKernelGGL(pp_split_planes, dim3((unsigned)(nProto / 4 / 256)), dim3(256),
                       0, stream, proto, phi, plo, (int)(nProto / 4));
    hipLaunchKernelGGL(pp_gemm_dist_planes, gemmGrid, dim3(256), 0, stream,
                       ehi, elo, phi, plo, x2, y2, out_dist, minbuf);
  } else {
    hipLaunchKernelGGL(pp_gemm_dist_conv, gemmGrid, dim3(256), 0, stream,
                       emb, proto, x2, y2, out_dist, minbuf);
  }

  hipLaunchKernelGGL(pp_finalize, dim3(B_), dim3(256), 0, stream,
                     minbuf, fc, out_pd, out_cls);
}
